// EdgeFeatureNet_69870527971631
// MI455X (gfx1250) — matrix operand order repaired and checked by reading, not compiled, not run on hardware
//
#include <hip/hip_runtime.h>
#include <hip/hip_bf16.h>

// ---------------------------------------------------------------------------
// EdgeFeatureNet fused for MI455X (gfx1250, wave32, WMMA 16x16x32 f16->f32)
//
// Roofline: 410MB fp32 output -> ~20us HBM floor @23.3TB/s; MLP ~88 GFLOP on
// v_wmma_f32_16x16x32_f16. Key balance knob: weight (B-fragment) cache
// traffic. Each wave owns M=32 edges (2 subtiles) so every B fragment feeds
// two WMMAs, and all A fragments for a layer are held in VGPRs (loaded once
// from LDS), keeping the shared per-WGP LDS out of the critical path.
// ---------------------------------------------------------------------------

typedef _Float16 h16;
typedef __attribute__((ext_vector_type(16))) _Float16 v16h;
typedef __attribute__((ext_vector_type(8)))  float    v8f;

#define C_S    256
#define FEAT   64
#define C_P    128
#define NBINS  22
#define K1PAD  192
#define K1REAL 172

// Per-wave LDS layout (bytes):
#define OFF_FEATS 0        // 32 rows * 400B  (192+8 halves, layer-1 A panel)
#define OFF_H     12800    // 32 rows * 272B  (layer-1 D / layer-2 A)
// layer-2 D / layer-3 A reuses OFF_FEATS with stride 272 (8704B)
#define OFF_OUT   8704     // 32 rows * 512B  f32 D staging (ends 25088)
#define OFF_STATS 25088    // 32 * {mean, rstd} f32 (256B)
#define OFF_GB    25344    // gamma[128], beta[128] f32 (1024B)
#define WAVE_LDS  26368
#define WAVES_PER_BLOCK 2

union Frag32B { uint4 u4[2]; v16h v; };

// ---------------------------------------------------------------------------
// Kernel 1: p_i = embed @ W_sp + b_sp, stored f16 [N][64]
// ---------------------------------------------------------------------------
__global__ void node_proj_kernel(const float* __restrict__ embed,
                                 const float* __restrict__ Wsp,
                                 const float* __restrict__ bsp,
                                 h16* __restrict__ p_out, int N) {
  __shared__ float row[4][C_S];
  const int rg  = threadIdx.x >> 6;
  const int col = threadIdx.x & 63;
  const int r = blockIdx.x * 4 + rg;
  const bool valid = (r < N);
  if (valid) {
    const float* er = embed + (size_t)r * C_S;
#pragma unroll
    for (int i = 0; i < 4; ++i) row[rg][col + i * 64] = er[col + i * 64];
  }
  __syncthreads();
  if (!valid) return;
  float acc = bsp[col];
#pragma unroll 8
  for (int k = 0; k < C_S; ++k) acc += row[rg][k] * Wsp[k * FEAT + col];
  p_out[(size_t)r * FEAT + col] = (h16)acc;
}

// ---------------------------------------------------------------------------
// Kernel 2: repack W1/W2/W3 (fp32 [K][128]) into B-fragment layout.
// B fragment (16x16x32): lane L holds column n = L%16, g=L/16; slots 0..7
// hold K=kt*32+8g+0..7, slots 8..15 hold K=kt*32+16+8g+0..7.
// Storage: ((kt*8+nt)*512 + lane*16 + s) halves -> 32B contiguous per lane.
// ---------------------------------------------------------------------------
__global__ void repack_kernel(const float* __restrict__ W1,
                              const float* __restrict__ W2,
                              const float* __restrict__ W3,
                              h16* __restrict__ wf) {
  const int L1 = 6 * 8 * 512;
  const int L2 = 4 * 8 * 512;
  int tid = blockIdx.x * blockDim.x + threadIdx.x;
  if (tid >= L1 + 2 * L2) return;
  const float* W; int Kreal; int base; int u;
  if (tid < L1)           { W = W1; Kreal = K1REAL; base = 0;       u = tid; }
  else if (tid < L1 + L2) { W = W2; Kreal = C_P;    base = L1;      u = tid - L1; }
  else                    { W = W3; Kreal = C_P;    base = L1 + L2; u = tid - L1 - L2; }
  const int frag   = u >> 9;
  const int within = u & 511;
  const int lane = within >> 4, s = within & 15;
  const int kt = frag >> 3, nt = frag & 7;
  const int k = kt * 32 + 8 * (lane >> 4) + (s & 7) + 16 * (s >> 3);
  const int n = nt * 16 + (lane & 15);
  const float v = (k < Kreal) ? W[k * C_P + n] : 0.0f;
  wf[base + u] = (h16)v;
}

// ---------------------------------------------------------------------------
// Kernel 3: fused edge pipeline. One wave = 32 edges (2 x 16-row subtiles).
// A-fragment pattern (ISA 7.12.2, 16-bit A 16x32): lane m=L%16, g=L/16:
//   slots 0..7  -> K = kt*32 + 8g + 0..7        (aligned b128)
//   slots 8..15 -> K = kt*32 + 16 + 8g + 0..7   (aligned b128)
// ---------------------------------------------------------------------------
__global__ void __launch_bounds__(WAVES_PER_BLOCK * 32) edge_kernel(
    const h16*  __restrict__ p_i,
    const float* __restrict__ trans_t,
    const float* __restrict__ trans_sc,
    const int*  __restrict__ eidx,
    const h16*  __restrict__ w1f,
    const float* __restrict__ b1,
    const h16*  __restrict__ w2f,
    const float* __restrict__ b2,
    const h16*  __restrict__ w3f,
    const float* __restrict__ b3,
    const float* __restrict__ gamma,
    const float* __restrict__ beta,
    float* __restrict__ out,
    int E, int ntiles) {
  __shared__ __align__(16) unsigned char smem[WAVES_PER_BLOCK * WAVE_LDS];
  const int lane = threadIdx.x & 31;
  const int wv   = threadIdx.x >> 5;
  const int tile = blockIdx.x * WAVES_PER_BLOCK + wv;   // 32-edge tile
  if (tile >= ntiles) return;

  unsigned char* wbase = smem + wv * WAVE_LDS;
  const int el = lane & 15;   // row-in-subtile / column-in-ntile
  const int g  = lane >> 4;   // lane half-group

  // ---- Phase 0: preload gamma/beta, gather features, distogram -----------
  {
    float* gb = (float*)(wbase + OFF_GB);
    for (int i = lane; i < 2 * C_P; i += 32)
      gb[i] = (i < C_P) ? gamma[i] : beta[i - C_P];
  }
#pragma unroll
  for (int sub = 0; sub < 2; ++sub) {
    int edge = tile * 32 + sub * 16 + el;
    if (edge >= E) edge = E - 1;
    const int src = eidx[edge];
    const int tgt = eidx[E + edge];
    unsigned char* frow = wbase + OFF_FEATS + (sub * 16 + el) * 400;
    {
      const int node = (g == 0) ? tgt : src;  // feats[0:64]=p_i[tgt], [64:128]=p_i[src]
      const uint4* pp = (const uint4*)(p_i + (size_t)node * FEAT);
      uint4* dst = (uint4*)(frow + g * 128);
#pragma unroll
      for (int i = 0; i < 8; ++i) dst[i] = pp[i];
    }
    {
      const float* tr = (g == 0) ? trans_t : trans_sc;
      const float dx = tr[tgt * 3 + 0] - tr[src * 3 + 0];
      const float dy = tr[tgt * 3 + 1] - tr[src * 3 + 1];
      const float dz = tr[tgt * 3 + 2] - tr[src * 3 + 2];
      const float d = sqrtf(dx * dx + dy * dy + dz * dz);
      const float step = (20.0f - 0.001f) / 21.0f;
      h16* brow = (h16*)frow + 128 + g * NBINS;
#pragma unroll
      for (int i = 0; i < NBINS; ++i) {
        const float lo = 0.001f + i * step;
        const float hi = (i < NBINS - 1) ? (0.001f + (i + 1) * step) : 1.0e8f;
        brow[i] = (h16)((d > lo && d < hi) ? 1.0f : 0.0f);
      }
      if (g == 0) {
        h16* z = (h16*)frow + K1REAL;
#pragma unroll
        for (int i = 0; i < K1PAD - K1REAL; ++i) z[i] = (h16)0.0f;
      }
    }
  }

  auto loadA = [&](int baseOff, int strideB, int sub, int kt) -> v16h {
    Frag32B a;
    const unsigned char* rp =
        wbase + baseOff + (sub * 16 + el) * strideB + kt * 64 + 16 * g;
    a.u4[0] = *(const uint4*)rp;
    a.u4[1] = *(const uint4*)(rp + 32);
    return a.v;
  };
  auto loadB = [&](const h16* wf, int kt, int nt) -> v16h {
    Frag32B b;
    const uint4* wp = (const uint4*)(wf + ((size_t)(kt * 8 + nt) * 512) + lane * 16);
    b.u4[0] = wp[0];
    b.u4[1] = wp[1];
    return b.v;
  };

  // ---- Layer 1: [32x192] @ W1 -> relu -> OFF_H (stride 272) --------------
  {
    v16h aF[2][6];
#pragma unroll
    for (int sub = 0; sub < 2; ++sub)
#pragma unroll
      for (int kt = 0; kt < 6; ++kt) aF[sub][kt] = loadA(OFF_FEATS, 400, sub, kt);
    h16* hp = (h16*)(wbase + OFF_H);
#pragma unroll
    for (int nt = 0; nt < 8; ++nt) {
      const float bv = b1[nt * 16 + el];
      v8f c0, c1;
#pragma unroll
      for (int j = 0; j < 8; ++j) { c0[j] = bv; c1[j] = bv; }
#pragma unroll
      for (int kt = 0; kt < 6; ++kt) {
        v16h b = loadB(w1f, kt, nt);
        c0 = __builtin_amdgcn_wmma_f32_16x16x32_f16(false, aF[0][kt], false, b, (short)0, c0, false, false);
        c1 = __builtin_amdgcn_wmma_f32_16x16x32_f16(false, aF[1][kt], false, b, (short)0, c1, false, false);
      }
#pragma unroll
      for (int j = 0; j < 8; ++j) {
        hp[(j + 8 * g) * 136 + nt * 16 + el]        = (h16)fmaxf(c0[j], 0.0f);
        hp[(16 + j + 8 * g) * 136 + nt * 16 + el]   = (h16)fmaxf(c1[j], 0.0f);
      }
    }
  }

  // ---- Layer 2: OFF_H @ W2 -> relu -> OFF_FEATS (stride 272) -------------
  {
    v16h aF[2][4];
#pragma unroll
    for (int sub = 0; sub < 2; ++sub)
#pragma unroll
      for (int kt = 0; kt < 4; ++kt) aF[sub][kt] = loadA(OFF_H, 272, sub, kt);
    h16* hp = (h16*)(wbase + OFF_FEATS);
#pragma unroll
    for (int nt = 0; nt < 8; ++nt) {
      const float bv = b2[nt * 16 + el];
      v8f c0, c1;
#pragma unroll
      for (int j = 0; j < 8; ++j) { c0[j] = bv; c1[j] = bv; }
#pragma unroll
      for (int kt = 0; kt < 4; ++kt) {
        v16h b = loadB(w2f, kt, nt);
        c0 = __builtin_amdgcn_wmma_f32_16x16x32_f16(false, aF[0][kt], false, b, (short)0, c0, false, false);
        c1 = __builtin_amdgcn_wmma_f32_16x16x32_f16(false, aF[1][kt], false, b, (short)0, c1, false, false);
      }
#pragma unroll
      for (int j = 0; j < 8; ++j) {
        hp[(j + 8 * g) * 136 + nt * 16 + el]        = (h16)fmaxf(c0[j], 0.0f);
        hp[(16 + j + 8 * g) * 136 + nt * 16 + el]   = (h16)fmaxf(c1[j], 0.0f);
      }
    }
  }

  // ---- Layer 3: OFF_FEATS @ W3 -> f32 D staging + running LN stats -------
  float s0[8], q0[8], s1[8], q1[8];
#pragma unroll
  for (int j = 0; j < 8; ++j) { s0[j] = q0[j] = s1[j] = q1[j] = 0.0f; }
  {
    v16h aF[2][4];
#pragma unroll
    for (int sub = 0; sub < 2; ++sub)
#pragma unroll
      for (int kt = 0; kt < 4; ++kt) aF[sub][kt] = loadA(OFF_FEATS, 272, sub, kt);
    float* dp = (float*)(wbase + OFF_OUT);
#pragma unroll
    for (int nt = 0; nt < 8; ++nt) {
      const float bv = b3[nt * 16 + el];
      v8f c0, c1;
#pragma unroll
      for (int j = 0; j < 8; ++j) { c0[j] = bv; c1[j] = bv; }
#pragma unroll
      for (int kt = 0; kt < 4; ++kt) {
        v16h b = loadB(w3f, kt, nt);
        c0 = __builtin_amdgcn_wmma_f32_16x16x32_f16(false, aF[0][kt], false, b, (short)0, c0, false, false);
        c1 = __builtin_amdgcn_wmma_f32_16x16x32_f16(false, aF[1][kt], false, b, (short)0, c1, false, false);
      }
#pragma unroll
      for (int j = 0; j < 8; ++j) {
        const float v0 = c0[j], v1 = c1[j];
        s0[j] += v0; q0[j] += v0 * v0;
        s1[j] += v1; q1[j] += v1 * v1;
        dp[(j + 8 * g) * C_P + nt * 16 + el]      = v0;
        dp[(16 + j + 8 * g) * C_P + nt * 16 + el] = v1;
      }
    }
  }

  // ---- LayerNorm stats: reduce within 16-lane halves, publish via LDS ----
  {
    float2* st = (float2*)(wbase + OFF_STATS);
#pragma unroll
    for (int j = 0; j < 8; ++j) {
      float a = s0[j], b = q0[j], c = s1[j], d = q1[j];
#pragma unroll
      for (int m = 8; m >= 1; m >>= 1) {
        a += __shfl_xor(a, m, 32);
        b += __shfl_xor(b, m, 32);
        c += __shfl_xor(c, m, 32);
        d += __shfl_xor(d, m, 32);
      }
      if (el == 0) {
        float mean = a * (1.0f / 128.0f);
        float var  = b * (1.0f / 128.0f) - mean * mean;
        st[j + 8 * g]      = make_float2(mean, rsqrtf(var + 1e-5f));
        mean = c * (1.0f / 128.0f);
        var  = d * (1.0f / 128.0f) - mean * mean;
        st[16 + j + 8 * g] = make_float2(mean, rsqrtf(var + 1e-5f));
      }
    }
  }

  // ---- Normalize + coalesced b128 stores ---------------------------------
  {
    const uint4* dp4 = (const uint4*)(wbase + OFF_OUT);
    const float2* st = (const float2*)(wbase + OFF_STATS);
    const float* gb  = (const float*)(wbase + OFF_GB);
    for (int i = lane; i < 32 * (C_P / 4); i += 32) {
      const int row = i >> 5;               // 32 uint4 per row
      const int gedge = tile * 32 + row;
      if (gedge >= E) break;                // rows are in ascending order
      const int c0 = (i & 31) * 4;
      uint4 du = dp4[i];
      float4 d = *(float4*)&du;
      const float2 ms = st[row];
      float4 o;
      o.x = (d.x - ms.x) * ms.y * gb[c0 + 0] + gb[C_P + c0 + 0];
      o.y = (d.y - ms.x) * ms.y * gb[c0 + 1] + gb[C_P + c0 + 1];
      o.z = (d.z - ms.x) * ms.y * gb[c0 + 2] + gb[C_P + c0 + 2];
      o.w = (d.w - ms.x) * ms.y * gb[c0 + 3] + gb[C_P + c0 + 3];
      *(uint4*)(out + (size_t)gedge * C_P + c0) = *(uint4*)&o;
    }
  }
}

// ---------------------------------------------------------------------------
extern "C" void kernel_launch(void* const* d_in, const int* in_sizes, int n_in,
                              void* d_out, int out_size, void* d_ws, size_t ws_size,
                              hipStream_t stream) {
  const float* embed    = (const float*)d_in[0];
  const float* trans_t  = (const float*)d_in[1];
  const float* trans_sc = (const float*)d_in[2];
  const float* W_sp     = (const float*)d_in[3];
  const float* b_sp     = (const float*)d_in[4];
  const float* W1       = (const float*)d_in[5];
  const float* b1       = (const float*)d_in[6];
  const float* W2       = (const float*)d_in[7];
  const float* b2       = (const float*)d_in[8];
  const float* W3       = (const float*)d_in[9];
  const float* b3       = (const float*)d_in[10];
  const float* ln_g     = (const float*)d_in[11];
  const float* ln_b     = (const float*)d_in[12];
  const int*   eidx     = (const int*)d_in[13];
  float* out = (float*)d_out;

  const int N = in_sizes[0] / C_S;
  const int E = in_sizes[13] / 2;

  h16* p_i = (h16*)d_ws;
  const size_t piBytes = (((size_t)N * FEAT * 2) + 255) & ~(size_t)255;
  h16* wfrag = (h16*)((char*)d_ws + piBytes);
  h16* w1f = wfrag;
  h16* w2f = wfrag + 6 * 8 * 512;
  h16* w3f = wfrag + 6 * 8 * 512 + 4 * 8 * 512;

  node_proj_kernel<<<(N + 3) / 4, 256, 0, stream>>>(embed, W_sp, b_sp, p_i, N);

  const int repackThreads = (6 + 4 + 4) * 8 * 512;
  repack_kernel<<<(repackThreads + 255) / 256, 256, 0, stream>>>(W1, W2, W3, wfrag);

  const int ntiles = (E + 31) / 32;
  edge_kernel<<<(ntiles + WAVES_PER_BLOCK - 1) / WAVES_PER_BLOCK,
                WAVES_PER_BLOCK * 32, 0, stream>>>(
      p_i, trans_t, trans_sc, eidx, w1f, b1, w2f, b2, w3f, b3, ln_g, ln_b,
      out, E, ntiles);
}
